// GCN_with_emb_15556371546266
// MI455X (gfx1250) — matrix operand order repaired and checked by reading, not compiled, not run on hardware
//
#include <hip/hip_runtime.h>

// ---------------------------------------------------------------------------
// GCN (2-layer) for MI455X / gfx1250.
// Dense GEMMs via v_wmma_f32_16x16x32_bf16 (bf16 in, f32 accumulate).
// Edge aggregation via global_atomic_add_f32 (memory-bound part).
// ---------------------------------------------------------------------------

#define N_NODES 50000
#define F_IN    512
#define H_DIM   256
#define C_DIM   64

typedef __attribute__((ext_vector_type(16))) __bf16 v16bf;
typedef __attribute__((ext_vector_type(8)))  float  v8f;

__device__ __forceinline__ unsigned short f32_to_bf16_rne(float f) {
    unsigned int u = __float_as_uint(f);
    unsigned int lsb = (u >> 16) & 1u;
    u += 0x7fffu + lsb;                 // round to nearest even
    return (unsigned short)(u >> 16);
}

// ---- elementwise f32 -> bf16 ------------------------------------------------
__global__ __launch_bounds__(256) void cvt_bf16_kernel(
    const float* __restrict__ in, unsigned short* __restrict__ out, int n) {
    int i = blockIdx.x * blockDim.x + threadIdx.x;
    if (i < n) out[i] = f32_to_bf16_rne(in[i]);
}

// ---- pack weight [K, NCOL] f32 -> dwords {W[2k,n] | W[2k+1,n]<<16} ----------
__global__ __launch_bounds__(256) void pack_w_kernel(
    const float* __restrict__ W, unsigned int* __restrict__ Wp, int K, int NCOL) {
    int idx = blockIdx.x * blockDim.x + threadIdx.x;
    int total = (K >> 1) * NCOL;
    if (idx >= total) return;
    int n  = idx % NCOL;
    int k2 = idx / NCOL;
    unsigned int lo = f32_to_bf16_rne(W[(size_t)(2 * k2)     * NCOL + n]);
    unsigned int hi = f32_to_bf16_rne(W[(size_t)(2 * k2 + 1) * NCOL + n]);
    Wp[idx] = lo | (hi << 16);
}

// ---- dense GEMM: C[M,NCOL] = A[M,K](bf16) x B[K,NCOL](packed bf16) ----------
// One wave per 16x16 output tile. K % 32 == 0, NCOL % 16 == 0.
template <int K, int NCOL>
__global__ __launch_bounds__(256) void gemm_wmma_bf16(
    const unsigned short* __restrict__ A,     // [M,K] bf16 row-major
    const unsigned int*   __restrict__ Bpack, // [K/2, NCOL] dwords
    float* __restrict__ C, int Mtiles) {
    const int NT   = NCOL / 16;
    int wave = blockIdx.x * (blockDim.x >> 5) + (threadIdx.x >> 5);
    if (wave >= Mtiles * NT) return;          // wave-uniform: EXEC stays all-1s
    int lane  = threadIdx.x & 31;
    int tileM = wave / NT;
    int tileN = wave - tileM * NT;
    int hi    = lane >> 4;                    // half-wave select
    int l15   = lane & 15;

    const unsigned short* Arow = A + (size_t)(tileM * 16 + l15) * K;
    int ncol = tileN * 16 + l15;

    v8f acc = {};
    for (int k0 = 0; k0 < K; k0 += 32) {
        union { v16bf v; unsigned int u[8]; } a, b;
        // A 16x32 bf16 layout: lanes 0-15 rows M, K-halves 0-7/16-23;
        // lanes 16-31 same rows, K-halves 8-15/24-31.
        int ka = k0 + hi * 8;
#pragma unroll
        for (int v = 0; v < 4; ++v) {
            a.u[v]     = *(const unsigned int*)(Arow + ka + 2 * v);
            a.u[v + 4] = *(const unsigned int*)(Arow + ka + 16 + 2 * v);
        }
        // B 32x16 layout: lanes 0-15 hold K=k0+0..7/16-23, lanes 16-31 K=8-15/24-31;
        // each VGPR = 2 consecutive K packed in one dword, column = lane&15.
        int kb2 = (k0 >> 1) + hi * 4;
#pragma unroll
        for (int v = 0; v < 8; ++v)
            b.u[v] = Bpack[(size_t)(kb2 + v + (v & 4)) * NCOL + ncol];

        acc = __builtin_amdgcn_wmma_f32_16x16x32_bf16(
            /*neg_a=*/false, a.v, /*neg_b=*/false, b.v,
            /*c_mod=*/(short)0, acc, /*reuse_a=*/false, /*reuse_b=*/false);
    }
    // D layout: VGPR i -> row i (lanes 0-15) / i+8 (lanes 16-31), col = lane&15
    float* Cr = C + (size_t)(tileM * 16) * NCOL + tileN * 16 + l15;
#pragma unroll
    for (int i = 0; i < 8; ++i)
        Cr[(size_t)(i + hi * 8) * NCOL] = acc[i];
}

// ---- edge scatter-add: agg[dst] += sup[src] * w -----------------------------
__global__ __launch_bounds__(256) void scatter_add_kernel(
    const float* __restrict__ sup, const int* __restrict__ src,
    const int* __restrict__ dst, const float* __restrict__ w,
    float* __restrict__ agg, int shift, int mask, int total) {
    int idx = blockIdx.x * blockDim.x + threadIdx.x;
    if (idx >= total) return;
    int e = idx >> shift;
    int f = idx & mask;
    float val = sup[((size_t)src[e] << shift) + f] * w[e];
    atomicAdd(&agg[((size_t)dst[e] << shift) + f], val);
}

// ---- relu -> emb (f32) + bf16 copy for next GEMM ---------------------------
__global__ __launch_bounds__(256) void relu_emb_cvt_kernel(
    const float* __restrict__ agg, float* __restrict__ emb,
    unsigned short* __restrict__ hbf, int n) {
    int i = blockIdx.x * blockDim.x + threadIdx.x;
    if (i >= n) return;
    float v = fmaxf(agg[i], 0.0f);
    emb[i] = v;
    hbf[i] = f32_to_bf16_rne(v);
}

// ---- relu + log_softmax over C=64, one wave per node ------------------------
__global__ __launch_bounds__(256) void relu_logsoftmax_kernel(
    const float* __restrict__ agg, float* __restrict__ out, int n) {
    int wave = blockIdx.x * (blockDim.x >> 5) + (threadIdx.x >> 5);
    int lane = threadIdx.x & 31;
    if (wave >= n) return;
    const float* r = agg + (size_t)wave * C_DIM;
    float v0 = fmaxf(r[lane], 0.0f);
    float v1 = fmaxf(r[lane + 32], 0.0f);
    float m = fmaxf(v0, v1);
#pragma unroll
    for (int off = 16; off > 0; off >>= 1) m = fmaxf(m, __shfl_xor(m, off, 32));
    float s = __expf(v0 - m) + __expf(v1 - m);
#pragma unroll
    for (int off = 16; off > 0; off >>= 1) s += __shfl_xor(s, off, 32);
    float lse = m + __logf(s);
    float* o = out + (size_t)wave * C_DIM;
    o[lane]      = v0 - lse;
    o[lane + 32] = v1 - lse;
}

extern "C" void kernel_launch(void* const* d_in, const int* in_sizes, int n_in,
                              void* d_out, int out_size, void* d_ws, size_t ws_size,
                              hipStream_t stream) {
    (void)n_in; (void)out_size; (void)ws_size;
    const float* x    = (const float*)d_in[0];   // [N, 512]
    const int*   esrc = (const int*)  d_in[1];   // [E]
    const int*   edst = (const int*)  d_in[2];   // [E]
    const float* ew   = (const float*)d_in[3];   // [E]
    const float* W1   = (const float*)d_in[4];   // [512, 256]
    const float* W2   = (const float*)d_in[5];   // [256, 64]

    const int N = in_sizes[0] / F_IN;            // 50000
    const int E = in_sizes[1];                   // 1600000

    float* logits = (float*)d_out;               // [N, 64]
    float* emb    = (float*)d_out + (size_t)N * C_DIM;  // [N, 256]

    // ---- workspace layout (bytes, 256-aligned), with reuse ----
    char* ws = (char*)d_ws;
    size_t off = 0;
    unsigned short* xbf  = (unsigned short*)(ws + off); off += (size_t)N * F_IN * 2;   // 51.2 MB
    unsigned int*   w1p  = (unsigned int*)  (ws + off); off += (size_t)(F_IN/2) * H_DIM * 4; // 256 KB
    unsigned int*   w2p  = (unsigned int*)  (ws + off); off += (size_t)(H_DIM/2) * C_DIM * 4; // 32 KB
    off = (off + 255) & ~(size_t)255;
    float*          sup1 = (float*)(ws + off);          // 51.2 MB (reused as sup2)
    float*          sup2 = sup1;
    off += (size_t)N * H_DIM * 4;
    float*          agg1 = (float*)(ws + off);          // 51.2 MB (reused as agg2)
    float*          agg2 = agg1;
    off += (size_t)N * H_DIM * 4;
    unsigned short* hbf  = (unsigned short*)(ws + off); // 25.6 MB

    // 1) convert x to bf16; pack W1, W2
    {
        int n = N * F_IN;
        cvt_bf16_kernel<<<(n + 255) / 256, 256, 0, stream>>>(x, xbf, n);
        int t1 = (F_IN / 2) * H_DIM;
        pack_w_kernel<<<(t1 + 255) / 256, 256, 0, stream>>>(W1, w1p, F_IN, H_DIM);
        int t2 = (H_DIM / 2) * C_DIM;
        pack_w_kernel<<<(t2 + 255) / 256, 256, 0, stream>>>(W2, w2p, H_DIM, C_DIM);
    }

    const int Mtiles = N / 16;  // 3125 (exact)

    // 2) support1 = x @ W1  (WMMA bf16 -> f32)
    {
        int waves = Mtiles * (H_DIM / 16);
        gemm_wmma_bf16<F_IN, H_DIM><<<(waves + 7) / 8, 256, 0, stream>>>(xbf, w1p, sup1, Mtiles);
    }

    // 3) agg1 = segment_sum(support1[src] * w, dst)
    hipMemsetAsync(agg1, 0, (size_t)N * H_DIM * 4, stream);
    {
        int total = E * H_DIM;  // 4.096e8
        scatter_add_kernel<<<(total + 255) / 256, 256, 0, stream>>>(
            sup1, esrc, edst, ew, agg1, 8, H_DIM - 1, total);
    }

    // 4) h = relu(agg1) -> emb (f32 out) + hbf (bf16)
    {
        int n = N * H_DIM;
        relu_emb_cvt_kernel<<<(n + 255) / 256, 256, 0, stream>>>(agg1, emb, hbf, n);
    }

    // 5) support2 = h @ W2
    {
        int waves = Mtiles * (C_DIM / 16);
        gemm_wmma_bf16<H_DIM, C_DIM><<<(waves + 7) / 8, 256, 0, stream>>>(hbf, w2p, sup2, Mtiles);
    }

    // 6) agg2 = segment_sum(support2[src] * w, dst)
    hipMemsetAsync(agg2, 0, (size_t)N * C_DIM * 4, stream);
    {
        int total = E * C_DIM;  // 1.024e8
        scatter_add_kernel<<<(total + 255) / 256, 256, 0, stream>>>(
            sup2, esrc, edst, ew, agg2, 6, C_DIM - 1, total);
    }

    // 7) logits = log_softmax(relu(agg2))
    relu_logsoftmax_kernel<<<(N + 7) / 8, 256, 0, stream>>>(agg2, logits, N);
}
